// CrossAttention_3126736191504
// MI455X (gfx1250) — matrix operand order repaired and checked by reading, not compiled, not run on hardware
//
#include <hip/hip_runtime.h>
#include <math.h>
#include <stdint.h>

// ---------------------------------------------------------------------------
// Types
// ---------------------------------------------------------------------------
typedef __attribute__((ext_vector_type(16))) __bf16 v16bf;
typedef __attribute__((ext_vector_type(8)))  __bf16 bf8;
typedef __attribute__((ext_vector_type(4)))  __bf16 bf4;
typedef __attribute__((ext_vector_type(8)))  float  v8f;

union V16 { v16bf v; bf8 h[2]; };

__device__ __forceinline__ __bf16 f2bf(float x) {
  union { float f; unsigned u; } a; a.f = x;
  unsigned r = a.u + 0x7FFFu + ((a.u >> 16) & 1u);   // round-to-nearest-even
  unsigned short hs = (unsigned short)(r >> 16);
  union { unsigned short u; __bf16 b; } c; c.u = hs;
  return c.b;
}

// Problem constants
#define NTOK 8192      // B * L = 4 * 2048
#define DEMB 1024
#define NHEADS 16
#define DK 64

// ---------------------------------------------------------------------------
// Tensor Data Mover support (CDNA5 TDM, tracked by TENSORcnt)
// ---------------------------------------------------------------------------
#if __has_builtin(__builtin_amdgcn_tensor_load_to_lds) && \
    __has_builtin(__builtin_amdgcn_s_wait_tensorcnt)
#define USE_TDM 1
#else
#define USE_TDM 0
#endif

#if __has_include(<hip/amd_detail/amd_gfx1250_TDM.h>)
#define TDM_SIX_ARG 1     // therock-10.0 headers -> 6-arg builtin
#else
#define TDM_SIX_ARG 0     // ROCm 7.2 -> 5-arg builtin
#endif

#if USE_TDM
typedef __attribute__((ext_vector_type(4))) unsigned int u32x4;
typedef __attribute__((ext_vector_type(8))) int          i32x8;
typedef __attribute__((ext_vector_type(4))) int          i32x4;

// LDS aperture generic address: low 32 bits are the LDS byte offset.
__device__ __forceinline__ unsigned lds_byte_off(const void* p) {
  return (unsigned)(unsigned long long)(uintptr_t)p;
}

// Issue a TDM 2D tile load: `rows` rows of 32 bf16 (64 B) from a row-major
// bf16 matrix with row stride DEMB elements, into LDS at lds_off with a
// +16 B pad after every 64 B row (so LDS row stride = 80 B = GEMM_LDSK elems).
__device__ __forceinline__ void tdm_load_tile(const __bf16* gsrc,
                                              unsigned lds_off, int rows) {
  unsigned long long ga = (unsigned long long)(uintptr_t)gsrc;
  u32x4 g0;
  g0[0] = 1u;                                        // count=1, user mode
  g0[1] = lds_off;                                   // lds_addr (bytes)
  g0[2] = (unsigned)(ga & 0xFFFFFFFFu);              // global_addr[31:0]
  g0[3] = (unsigned)((ga >> 32) & 0x01FFFFFFu)       // global_addr[56:32]
          | 0x80000000u;                             // type=2 ("image")
  i32x8 g1;
  // mask=0 | data_size=1 (2B) | pad_enable | pad_interval=3 (16 DW = 64 B)
  // | pad_amount=3 (4 DW = 16 B)
  g1[0] = (int)((1u << 16) | (1u << 20) | (3u << 22) | (3u << 25));
  g1[1] = 0;                                // abar addr = 0, tensor_dim0[15:0]=0
  g1[2] = (int)0x00004000u;                 // tensor_dim0=0x40000000 (no clip)
  g1[3] = (int)((32u << 16) | 0x4000u);     // tensor_dim1=0x40000000, tile_dim0=32
  g1[4] = rows;                             // tile_dim1=rows, tile_dim2=0
  g1[5] = DEMB;                             // tensor_dim0_stride = 1024 elems
  g1[6] = 0;                                // stride hi / dim1_stride lo
  g1[7] = 0;
  i32x4 z4 = (i32x4){0, 0, 0, 0};
#if TDM_SIX_ARG
  i32x8 z8 = (i32x8){0, 0, 0, 0, 0, 0, 0, 0};
  __builtin_amdgcn_tensor_load_to_lds(g0, g1, z4, z4, z8, 0);
#else
  __builtin_amdgcn_tensor_load_to_lds(g0, g1, z4, z4, 0);
#endif
}
#endif  // USE_TDM

// ---------------------------------------------------------------------------
// Kernel 1: f32 -> bf16 conversion (vectorized x4)
// ---------------------------------------------------------------------------
__global__ void cvt_f32_to_bf16(const float* __restrict__ src,
                                __bf16* __restrict__ dst, int n4) {
  int i = blockIdx.x * blockDim.x + threadIdx.x;
  if (i >= n4) return;
  float4 f = ((const float4*)src)[i];
  bf4 o;
  o[0] = f2bf(f.x); o[1] = f2bf(f.y); o[2] = f2bf(f.z); o[3] = f2bf(f.w);
  *(bf4*)(dst + (size_t)i * 4) = o;
}

// ---------------------------------------------------------------------------
// Kernel 2: NT GEMM, C[m,n] = sum_k A[m,k] * W[n,k]
// A: [M, 1024] bf16 row-major,  W: [N, 1024] bf16 row-major.
// WG = 256 threads = 8 waves (2x4), WG tile 128x256, BK = 32.
// Each wave: 64x64 output via 4x4 v_wmma_f32_16x16x32_bf16 accumulators.
// LDS tiles double-buffered and filled by the Tensor Data Mover; the main
// loop is manually unrolled 2x to match the double-buffer period so each
// phase uses compile-time buffer pointers (one s_wait_tensorcnt + one
// barrier per K-step, TDM of step k+1 overlapped with WMMA of step k).
// ---------------------------------------------------------------------------
#define GEMM_LDSK 40   // 32 + 8 bf16 pad -> 80 B row stride (TDM pad produces this)

template <bool STORE_F32>
__global__ void __launch_bounds__(256)
gemm_nt_bf16(const __bf16* __restrict__ A, const __bf16* __restrict__ W,
             void* __restrict__ C) {
  __shared__ __bf16 sA[2][128 * GEMM_LDSK];   // 2 x 10,240 B
  __shared__ __bf16 sB[2][256 * GEMM_LDSK];   // 2 x 20,480 B

  const int tid  = threadIdx.x;
  const int lane = tid & 31;
  const int wid  = tid >> 5;
  const int waveRow = wid >> 2;       // 0..1
  const int waveCol = wid & 3;        // 0..3
  const int rowBase = blockIdx.y * 128;
  const int colBase = blockIdx.x * 256;
  const int lm    = lane & 15;
  const int khalf = lane >> 4;

  v8f acc[4][4];
#pragma unroll
  for (int mi = 0; mi < 4; ++mi)
#pragma unroll
    for (int ni = 0; ni < 4; ++ni) acc[mi][ni] = (v8f){0,0,0,0,0,0,0,0};

  // One K-step: sync on the DMA filling (a0,b0), kick the DMA for nextK into
  // (na,nb), then run the 4x4 WMMA block out of (a0,b0).
  auto kstep = [&](int curK, const __bf16* a0, const __bf16* b0,
                   __bf16* na, __bf16* nb, int nextK, bool doNext) {
#if USE_TDM
    (void)curK;
    if (tid < 32) __builtin_amdgcn_s_wait_tensorcnt(0);
    __syncthreads();
    if (doNext && (tid < 32)) {
      tdm_load_tile(A + (size_t)rowBase * DEMB + nextK, lds_byte_off(na), 128);
      tdm_load_tile(W + (size_t)colBase * DEMB + nextK, lds_byte_off(nb), 256);
    }
#else
    (void)na; (void)nb; (void)nextK; (void)doNext;
    // Fallback: synchronous staging through VGPRs into (a0,b0).
#pragma unroll
    for (int i = 0; i < 2; ++i) {
      int c = tid + i * 256;
      int r = c >> 2, kc = (c & 3) << 3;
      bf8 vv = *(const bf8*)(A + (size_t)(rowBase + r) * DEMB + curK + kc);
      *(bf8*)((__bf16*)a0 + r * GEMM_LDSK + kc) = vv;
    }
#pragma unroll
    for (int i = 0; i < 4; ++i) {
      int c = tid + i * 256;
      int r = c >> 2, kc = (c & 3) << 3;
      bf8 vv = *(const bf8*)(W + (size_t)(colBase + r) * DEMB + curK + kc);
      *(bf8*)((__bf16*)b0 + r * GEMM_LDSK + kc) = vv;
    }
    __syncthreads();
#endif

    // A 16x32 bf16 layout: lane row = lm; elems 0..7 -> K = khalf*8+e,
    //                      elems 8..15 -> K = 16 + khalf*8 + e
    V16 af[4];
#pragma unroll
    for (int mi = 0; mi < 4; ++mi) {
      int r = waveRow * 64 + mi * 16 + lm;
      af[mi].h[0] = *(const bf8*)(a0 + r * GEMM_LDSK + khalf * 8);
      af[mi].h[1] = *(const bf8*)(a0 + r * GEMM_LDSK + 16 + khalf * 8);
    }
    // B 32x16 bf16 layout: lane col = lm; elems 0..7 -> K = khalf*8+e,
    //                      elems 8..15 -> K = 16 + khalf*8 + e
    V16 bfg[4];
#pragma unroll
    for (int ni = 0; ni < 4; ++ni) {
      int n = waveCol * 64 + ni * 16 + lm;
      bfg[ni].h[0] = *(const bf8*)(b0 + n * GEMM_LDSK + khalf * 8);
      bfg[ni].h[1] = *(const bf8*)(b0 + n * GEMM_LDSK + 16 + khalf * 8);
    }

#pragma unroll
    for (int mi = 0; mi < 4; ++mi)
#pragma unroll
      for (int ni = 0; ni < 4; ++ni)
        acc[mi][ni] = __builtin_amdgcn_wmma_f32_16x16x32_bf16(
            false, af[mi].v, false, bfg[ni].v, (short)0, acc[mi][ni],
            false, false);

#if !USE_TDM
    __syncthreads();
#endif
  };

#if USE_TDM
  // Prologue: wave 0 launches the first pair of tile DMAs into buffer 0.
  if (tid < 32) {
    tdm_load_tile(A + (size_t)rowBase * DEMB, lds_byte_off(&sA[0][0]), 128);
    tdm_load_tile(W + (size_t)colBase * DEMB, lds_byte_off(&sB[0][0]), 256);
  }
#endif

  // 2x-unrolled double-buffer pipeline: 16 pairs of K-steps cover K=1024.
  for (int k0 = 0; k0 < DEMB; k0 += 64) {
    kstep(k0,      &sA[0][0], &sB[0][0], &sA[1][0], &sB[1][0],
          k0 + 32, true);
    kstep(k0 + 32, &sA[1][0], &sB[1][0], &sA[0][0], &sB[0][0],
          k0 + 64, (k0 + 64) < DEMB);
  }

  // ---- epilogue: D layout row = khalf*8 + e, col = lm (per 16x16 tile) ----
#pragma unroll
  for (int mi = 0; mi < 4; ++mi) {
#pragma unroll
    for (int ni = 0; ni < 4; ++ni) {
      int row0 = rowBase + waveRow * 64 + mi * 16 + khalf * 8;
      int col  = colBase + waveCol * 64 + ni * 16 + lm;
#pragma unroll
      for (int e = 0; e < 8; ++e) {
        size_t idx = (size_t)(row0 + e) * DEMB + col;
        if (STORE_F32) ((float*)C)[idx] = acc[mi][ni][e];
        else           ((__bf16*)C)[idx] = f2bf(acc[mi][ni][e]);
      }
    }
  }
}

// ---------------------------------------------------------------------------
// Kernel 3: per-token attention. One wave per token, 4 tokens per WG.
//   Q_t[i,h] = q[h*64+i]  (64x16),  S = Q_t K_t^T / 8  (64x64, contract H=16)
//   P = softmax_rows(S),  O = P V_t (64x16),  concat[h*64+i] = O[i,h]
// QK / PV use bf16 WMMA with zero-padded contraction where needed.
// ---------------------------------------------------------------------------
__global__ void __launch_bounds__(128)
attn_kernel(const __bf16* __restrict__ Qp, const __bf16* __restrict__ Kp,
            const __bf16* __restrict__ Vp, __bf16* __restrict__ concat) {
  __shared__ float sS[4][64][65];   // 66,560 B

  const int lane  = threadIdx.x & 31;
  const int w     = threadIdx.x >> 5;
  const int token = blockIdx.x * 4 + w;
  const int lm    = lane & 15;
  const int khalf = lane >> 4;

  const __bf16* q = Qp + (size_t)token * DEMB;
  const __bf16* k = Kp + (size_t)token * DEMB;
  const __bf16* v = Vp + (size_t)token * DEMB;
  const __bf16 z = f2bf(0.0f);

  // ---- Q fragments (A operand): rows i = a*16+lm, contraction = h (0..15),
  //      K 16..31 zero-padded ----
  V16 aq[4];
#pragma unroll
  for (int a = 0; a < 4; ++a) {
    int i = a * 16 + lm;
#pragma unroll
    for (int e = 0; e < 8; ++e) {
      int h = khalf * 8 + e;                 // K = khalf*8 + e  (valid)
      aq[a].v[e]     = q[h * DK + i];
      aq[a].v[8 + e] = z;                    // K = 16 + ...     (pad)
    }
  }
  // ---- K fragments (B operand): cols j = b*16+lm; elems 0..7 hold K=h=khalf*8+e,
  //      elems 8..15 (K=16..31) are the zero pad ----
  V16 bk[4];
#pragma unroll
  for (int b = 0; b < 4; ++b) {
    int j = b * 16 + lm;
#pragma unroll
    for (int e = 0; e < 16; ++e)
      bk[b].v[e] = (e >= 8) ? z : k[(khalf * 8 + e) * DK + j];
  }

  // ---- S = Q K^T * 0.125, written to LDS ----
#pragma unroll
  for (int a = 0; a < 4; ++a) {
#pragma unroll
    for (int b = 0; b < 4; ++b) {
      v8f s = (v8f){0,0,0,0,0,0,0,0};
      s = __builtin_amdgcn_wmma_f32_16x16x32_bf16(false, aq[a].v, false,
                                                  bk[b].v, (short)0, s,
                                                  false, false);
#pragma unroll
      for (int e = 0; e < 8; ++e)
        sS[w][a * 16 + khalf * 8 + e][b * 16 + lm] = s[e] * 0.125f;
    }
  }
  __syncthreads();

  // ---- row softmax: each lane owns rows (lane) and (lane+32) ----
#pragma unroll
  for (int rr = 0; rr < 2; ++rr) {
    int r = lane + rr * 32;
    float m = -1e30f;
#pragma unroll 8
    for (int j = 0; j < 64; ++j) m = fmaxf(m, sS[w][r][j]);
    float sum = 0.0f;
#pragma unroll 8
    for (int j = 0; j < 64; ++j) {
      float e = __expf(sS[w][r][j] - m);
      sS[w][r][j] = e;
      sum += e;
    }
    float inv = 1.0f / sum;
#pragma unroll 8
    for (int j = 0; j < 64; ++j) sS[w][r][j] *= inv;
  }
  __syncthreads();

  // ---- V fragments (B operand): [K=32 of j][16 of h]; two j-steps ----
  V16 bv[2];
#pragma unroll
  for (int js = 0; js < 2; ++js) {
#pragma unroll
    for (int e = 0; e < 16; ++e) {
      int j = js * 32 + (e < 8 ? khalf * 8 + e : 16 + khalf * 8 + (e - 8));
      bv[js].v[e] = v[lm * DK + j];
    }
  }

  // ---- O = P V, then scatter to concat layout ----
#pragma unroll
  for (int a = 0; a < 4; ++a) {
    v8f o = (v8f){0,0,0,0,0,0,0,0};
#pragma unroll
    for (int js = 0; js < 2; ++js) {
      V16 ap;
      int i = a * 16 + lm;
#pragma unroll
      for (int e = 0; e < 8; ++e) {
        ap.v[e]     = f2bf(sS[w][i][js * 32 + khalf * 8 + e]);
        ap.v[8 + e] = f2bf(sS[w][i][js * 32 + 16 + khalf * 8 + e]);
      }
      o = __builtin_amdgcn_wmma_f32_16x16x32_bf16(false, ap.v, false,
                                                  bv[js].v, (short)0, o,
                                                  false, false);
    }
#pragma unroll
    for (int e = 0; e < 8; ++e) {
      int i = a * 16 + khalf * 8 + e;     // D row
      int h = lm;                         // D col
      concat[(size_t)token * DEMB + h * DK + i] = f2bf(o[e]);
    }
  }
}

// ---------------------------------------------------------------------------
// Host launch
// ---------------------------------------------------------------------------
extern "C" void kernel_launch(void* const* d_in, const int* in_sizes, int n_in,
                              void* d_out, int out_size, void* d_ws,
                              size_t ws_size, hipStream_t stream) {
  (void)in_sizes; (void)n_in; (void)out_size; (void)ws_size;

  const float* x   = (const float*)d_in[0];
  const float* ctx = (const float*)d_in[1];
  const float* Wq  = (const float*)d_in[2];
  const float* Wk  = (const float*)d_in[3];
  const float* Wv  = (const float*)d_in[4];
  const float* Wo  = (const float*)d_in[5];
  float* out = (float*)d_out;

  char* ws = (char*)d_ws;
  size_t off = 0;
  auto take = [&](size_t bytes) -> void* {
    void* p = ws + off;
    off = (off + bytes + 255) & ~(size_t)255;
    return p;
  };
  const size_t ACT = (size_t)NTOK * DEMB * sizeof(__bf16);  // 16 MiB
  const size_t WSZ = (size_t)DEMB * DEMB * sizeof(__bf16);  //  2 MiB
  __bf16* xb  = (__bf16*)take(ACT);
  __bf16* cb  = (__bf16*)take(ACT);
  __bf16* wqb = (__bf16*)take(WSZ);
  __bf16* wkb = (__bf16*)take(WSZ);
  __bf16* wvb = (__bf16*)take(WSZ);
  __bf16* wob = (__bf16*)take(WSZ);
  __bf16* Qp  = (__bf16*)take(ACT);
  __bf16* Kp  = (__bf16*)take(ACT);
  __bf16* Vp  = (__bf16*)take(ACT);
  __bf16* cc  = (__bf16*)take(ACT);

  // --- convert inputs to bf16 ---
  {
    int nAct4 = (int)((size_t)NTOK * DEMB / 4);
    int nW4   = (int)((size_t)DEMB * DEMB / 4);
    int tb = 256;
    cvt_f32_to_bf16<<<(nAct4 + tb - 1) / tb, tb, 0, stream>>>(x,   xb,  nAct4);
    cvt_f32_to_bf16<<<(nAct4 + tb - 1) / tb, tb, 0, stream>>>(ctx, cb,  nAct4);
    cvt_f32_to_bf16<<<(nW4   + tb - 1) / tb, tb, 0, stream>>>(Wq,  wqb, nW4);
    cvt_f32_to_bf16<<<(nW4   + tb - 1) / tb, tb, 0, stream>>>(Wk,  wkb, nW4);
    cvt_f32_to_bf16<<<(nW4   + tb - 1) / tb, tb, 0, stream>>>(Wv,  wvb, nW4);
    cvt_f32_to_bf16<<<(nW4   + tb - 1) / tb, tb, 0, stream>>>(Wo,  wob, nW4);
  }

  // --- QKV projections: [8192,1024] @ [1024,1024]^T (NT), bf16 out ---
  dim3 gemmGrid(DEMB / 256, NTOK / 128);   // (4, 64)
  gemm_nt_bf16<false><<<gemmGrid, 256, 0, stream>>>(xb, wqb, (void*)Qp);
  gemm_nt_bf16<false><<<gemmGrid, 256, 0, stream>>>(cb, wkb, (void*)Kp);
  gemm_nt_bf16<false><<<gemmGrid, 256, 0, stream>>>(cb, wvb, (void*)Vp);

  // --- per-token attention (64x64 over heads) ---
  attn_kernel<<<NTOK / 4, 128, 0, stream>>>(Qp, Kp, Vp, cc);

  // --- output projection: concat @ Wo^T, f32 out ---
  gemm_nt_bf16<true><<<gemmGrid, 256, 0, stream>>>(cc, wob, (void*)out);
}
